// QMixtralAttention_59554016526378
// MI455X (gfx1250) — compile-verified, matrix operand order checked by reading, not run on hardware
//
#include <hip/hip_runtime.h>
#include <hip/hip_bf16.h>

// Problem constants (match reference)
#define B_    2
#define S_    2048
#define HID_  4096
#define NH_   32
#define NKV_  8
#define HD_   128
#define P4_   3328
#define P6_   512
#define P8_   256

typedef __attribute__((ext_vector_type(16))) __bf16         v16bf;
typedef __attribute__((ext_vector_type(8)))  float          v8f;
typedef __attribute__((ext_vector_type(16))) unsigned short v16us;
typedef __attribute__((ext_vector_type(8)))  unsigned short v8us;
typedef __attribute__((ext_vector_type(4)))  unsigned short v4us;
typedef __attribute__((ext_vector_type(4)))  int            v4i;

// ---- async global->LDS staging (gfx1250 GLOBAL_LOAD_ASYNC_TO_LDS_B128) ----
// Guarded: host pass / toolchains without the builtin use a register-staged
// fallback identical to the previous (known-good) kernel.
#if defined(__has_builtin)
#if __has_builtin(__builtin_amdgcn_global_load_async_to_lds_b128)
#define HAVE_ASYNC_LDS 1
#endif
#endif

#ifdef HAVE_ASYNC_LDS
#define AS1_ __attribute__((address_space(1)))
#define AS3_ __attribute__((address_space(3)))
__device__ __forceinline__ void async_copy_b128(const unsigned short* g,
                                                unsigned short* l) {
  __builtin_amdgcn_global_load_async_to_lds_b128(
      (AS1_ v4i*)g, (AS3_ v4i*)l, 0, 0);
}
#if __has_builtin(__builtin_amdgcn_s_wait_asynccnt)
#define WAIT_ASYNC() __builtin_amdgcn_s_wait_asynccnt(0)
#else
#define WAIT_ASYNC() asm volatile("s_wait_asynccnt 0x0" ::: "memory")
#endif
#else
__device__ __forceinline__ void async_copy_b128(const unsigned short* g,
                                                unsigned short* l) {
  *(v8us*)l = *(const v8us*)g;
}
#define WAIT_ASYNC()
#endif

__device__ __forceinline__ unsigned short f2bf(float f) {
  unsigned int u = __builtin_bit_cast(unsigned int, f);
  u += 0x7FFFu + ((u >> 16) & 1u);          // round-to-nearest-even
  return (unsigned short)(u >> 16);
}
__device__ __forceinline__ float bf2f(unsigned short h) {
  unsigned int u = ((unsigned int)h) << 16;
  return __builtin_bit_cast(float, u);
}

// ---------------------------------------------------------------------------
// Kernel 1: gather by permutation + 3-group fake-quant -> bf16
// One block (256 threads) per row. Row stride == K == 4096 for all uses.
// ---------------------------------------------------------------------------
template <bool BF16IN>
__global__ void quant_gather(const void* __restrict__ xin,
                             const int*  __restrict__ idx,
                             unsigned short* __restrict__ xq,
                             int K)
{
  const int row = blockIdx.x;
  const int tid = threadIdx.x;
  const float*          xf = (const float*)xin;
  const unsigned short* xb = (const unsigned short*)xin;

  __shared__ float red[3][256];
  __shared__ float scale[3];

  float lmax[3] = {0.f, 0.f, 0.f};
  for (int c = tid; c < K; c += 256) {
    const int g   = (c < P4_) ? 0 : ((c < P4_ + P6_) ? 1 : 2);
    const int src = idx[c];
    const float v = BF16IN ? bf2f(xb[(size_t)row * K + src])
                           : xf[(size_t)row * K + src];
    lmax[g] = fmaxf(lmax[g], fabsf(v));
  }
  red[0][tid] = lmax[0]; red[1][tid] = lmax[1]; red[2][tid] = lmax[2];
  __syncthreads();
  for (int s = 128; s > 0; s >>= 1) {
    if (tid < s) {
      red[0][tid] = fmaxf(red[0][tid], red[0][tid + s]);
      red[1][tid] = fmaxf(red[1][tid], red[1][tid + s]);
      red[2][tid] = fmaxf(red[2][tid], red[2][tid + s]);
    }
    __syncthreads();
  }
  if (tid < 3) {
    const float qm[3] = {7.f, 31.f, 127.f};
    scale[tid] = red[tid][0] / qm[tid] + 1e-8f;
  }
  __syncthreads();

  for (int c = tid; c < K; c += 256) {
    const int   g  = (c < P4_) ? 0 : ((c < P4_ + P6_) ? 1 : 2);
    const float qm = (g == 0) ? 7.f : ((g == 1) ? 31.f : 127.f);
    const int src  = idx[c];
    const float v  = BF16IN ? bf2f(xb[(size_t)row * K + src])
                            : xf[(size_t)row * K + src];
    const float s  = scale[g];
    float q = rintf(v / s);
    q = fminf(fmaxf(q, -qm), qm);
    xq[(size_t)row * K + c] = f2bf(q * s);
  }
}

// ---------------------------------------------------------------------------
// Kernel 2: bf16 WMMA GEMM   Y[M x N] = A[M x K] * B[N x K]^T
// Block = 256 threads (8 waves). Tile 128x128, K-chunk 32.
// Double-buffered LDS with async global->LDS staging: the DMA for chunk
// k+1 is issued before the WMMAs of chunk k (no VGPR round-trip, tracked
// on ASYNCcnt), then s_wait_asynccnt + one barrier per chunk.
// ---------------------------------------------------------------------------
template <bool OUTF32>
__global__ __launch_bounds__(256)
void gemm_bf16(const unsigned short* __restrict__ A,
               const unsigned short* __restrict__ Bm,
               void* __restrict__ Y,
               int M, int N, int K)
{
  __shared__ __align__(32) unsigned short As[2][128 * 32];
  __shared__ __align__(32) unsigned short Bs[2][128 * 32];

  const int m0   = blockIdx.x * 128;
  const int n0   = blockIdx.y * 128;
  const int tid  = threadIdx.x;
  const int wave = tid >> 5;
  const int lane = tid & 31;
  const int lrow = lane & 15;
  const int lhalf = lane >> 4;

  v8f acc[8];
#pragma unroll
  for (int i = 0; i < 8; ++i)
    acc[i] = (v8f){0.f, 0.f, 0.f, 0.f, 0.f, 0.f, 0.f, 0.f};

  const int ldr = tid >> 1;          // 0..127
  const int ldc = (tid & 1) * 16;    // 0 or 16
  const unsigned short* arow = A  + (size_t)(m0 + ldr) * K + ldc;
  const unsigned short* brow = Bm + (size_t)(n0 + ldr) * K + ldc;
  unsigned short* asl0 = As[0] + ldr * 32 + ldc;
  unsigned short* bsl0 = Bs[0] + ldr * 32 + ldc;
  unsigned short* asl1 = As[1] + ldr * 32 + ldc;
  unsigned short* bsl1 = Bs[1] + ldr * 32 + ldc;

  // Prologue: stage chunk 0 into buffer 0 (16 bf16 per thread = 2 x b128)
  async_copy_b128(arow,     asl0);
  async_copy_b128(arow + 8, asl0 + 8);
  async_copy_b128(brow,     bsl0);
  async_copy_b128(brow + 8, bsl0 + 8);
  WAIT_ASYNC();
  __syncthreads();

  for (int k0 = 0; k0 < K; k0 += 32) {
    const int cur = (k0 >> 5) & 1;
    const bool more = (k0 + 32 < K);

    // Issue next chunk's async copies early (overlap with WMMAs below)
    if (more) {
      unsigned short* asn = cur ? asl0 : asl1;
      unsigned short* bsn = cur ? bsl0 : bsl1;
      async_copy_b128(arow + k0 + 32, asn);
      async_copy_b128(arow + k0 + 40, asn + 8);
      async_copy_b128(brow + k0 + 32, bsn);
      async_copy_b128(brow + k0 + 40, bsn + 8);
      __builtin_prefetch(arow + k0 + 64, 0, 1);
      __builtin_prefetch(brow + k0 + 64, 0, 1);
    }

    v16bf afrag = __builtin_bit_cast(
        v16bf, *(const v16us*)(As[cur] + (wave * 16 + lrow) * 32 + lhalf * 16));
#pragma unroll
    for (int nt = 0; nt < 8; ++nt) {
      v16bf bfrag = __builtin_bit_cast(
          v16bf, *(const v16us*)(Bs[cur] + (nt * 16 + lrow) * 32 + lhalf * 16));
      acc[nt] = __builtin_amdgcn_wmma_f32_16x16x32_bf16(
          false, afrag, false, bfrag, (short)0, acc[nt], false, false);
    }

    if (more) { WAIT_ASYNC(); }
    __syncthreads();
  }

#pragma unroll
  for (int nt = 0; nt < 8; ++nt) {
#pragma unroll
    for (int r = 0; r < 8; ++r) {
      const int m = m0 + wave * 16 + r + lhalf * 8;
      const int n = n0 + nt * 16 + lrow;
      const float v = acc[nt][r];
      if (OUTF32) ((float*)Y)[(size_t)m * N + n] = v;
      else        ((unsigned short*)Y)[(size_t)m * N + n] = f2bf(v);
    }
  }
}

// ---------------------------------------------------------------------------
// Kernel 3: RoPE on bf16 Q (32 heads) and K (8 heads), one block per (b,s)
// ---------------------------------------------------------------------------
__global__ void rope_kernel(unsigned short* __restrict__ Qb,
                            unsigned short* __restrict__ Kb,
                            const float* __restrict__ cosb,
                            const float* __restrict__ sinb)
{
  const int bs  = blockIdx.x;                 // b*S + s
  const int tid = threadIdx.x;
  for (int w = tid; w < (NH_ + NKV_) * (HD_ / 2); w += 256) {
    const int head = w >> 6;                  // HD/2 == 64
    const int d    = w & 63;
    const float c1 = cosb[(size_t)bs * HD_ + d];
    const float s1 = sinb[(size_t)bs * HD_ + d];
    const float c2 = cosb[(size_t)bs * HD_ + d + 64];
    const float s2 = sinb[(size_t)bs * HD_ + d + 64];
    unsigned short* base = (head < NH_)
        ? Qb + (size_t)bs * HID_ + (size_t)head * HD_
        : Kb + (size_t)bs * (NKV_ * HD_) + (size_t)(head - NH_) * HD_;
    const float x1 = bf2f(base[d]);
    const float x2 = bf2f(base[d + 64]);
    base[d]      = f2bf(x1 * c1 - x2 * s1);
    base[d + 64] = f2bf(x2 * c2 + x1 * s2);
  }
}

// ---------------------------------------------------------------------------
// Kernel 4: causal GQA flash attention, bf16 in/out, WMMA for QK^T and PV
// grid = (S/128, NH, B); block = 256 (8 waves); wave owns 16 q rows.
// Q tile lives in LDS to keep register pressure under the spill threshold.
// V^T staging uses a 4-key x 4-d per-thread micro-transpose so all DS
// stores are 8-byte (ds_store_b64 / 2addr_b64).
// ---------------------------------------------------------------------------
__global__ __launch_bounds__(256, 1)
void flash_kernel(const unsigned short* __restrict__ Q,
                  const unsigned short* __restrict__ Kc,
                  const unsigned short* __restrict__ Vc,
                  unsigned short* __restrict__ O)
{
  const int qt   = blockIdx.x;
  const int h    = blockIdx.y;
  const int b    = blockIdx.z;
  const int kv   = h >> 2;                    // NH/NKV = 4
  const int tid  = threadIdx.x;
  const int wave = tid >> 5;
  const int lane = tid & 31;
  const int lrow = lane & 15;
  const int lhalf = lane >> 4;

  __shared__ __align__(32) unsigned short Qs[8][16 * HD_];  // per-wave Q tile, 32KB
  __shared__ __align__(32) unsigned short Pt[8][16 * 32];   // per-wave P tile, 8KB
  __shared__ __align__(32) unsigned short Vts[HD_ * 32];    // shared V^T tile, 8KB
  unsigned short* P   = Pt[wave];
  unsigned short* Qsw = Qs[wave];

  const int qbase = qt * 128 + wave * 16;

  // Stage this wave's 16x128 Q tile into LDS (per-wave region; LDS ops are
  // in-order within a wave, so no barrier needed before reading it back).
  {
#pragma unroll
    for (int i = 0; i < 4; ++i) {
      const int c = lane * 4 + i;             // 128 chunks of 16 bf16
      const int m = c >> 3;                   // row 0..15
      const int p = (c & 7) * 16;             // 0..112
      v16us qv = *(const v16us*)(
          Q + (((size_t)b * S_ + qbase + m) * NH_ + h) * HD_ + p);
      *(v16us*)(Qsw + m * HD_ + p) = qv;
    }
  }

  float mrow[8], lsum[8];
#pragma unroll
  for (int r = 0; r < 8; ++r) { mrow[r] = -1e30f; lsum[r] = 0.f; }
  v8f acc[8];
#pragma unroll
  for (int i = 0; i < 8; ++i)
    acc[i] = (v8f){0.f, 0.f, 0.f, 0.f, 0.f, 0.f, 0.f, 0.f};

  const float scl = 0.08838834764831845f;     // HD^-0.5
  const int kend = qt * 128 + 128;            // uniform across block (causal)

  for (int ks = 0; ks < kend; ks += 32) {
    // ---- scores: two 16x16 tiles (keys ks..ks+15 and ks+16..ks+31)
    v8f c0 = (v8f){0.f,0.f,0.f,0.f,0.f,0.f,0.f,0.f};
    v8f c1 = c0;
    const unsigned short* kp0 =
        Kc + (((size_t)b * S_ + ks + lrow) * NKV_ + kv) * HD_;
    const unsigned short* kp1 =
        Kc + (((size_t)b * S_ + ks + 16 + lrow) * NKV_ + kv) * HD_;
#pragma unroll
    for (int ch = 0; ch < 4; ++ch) {
      v16bf qf = __builtin_bit_cast(
          v16bf, *(const v16us*)(Qsw + lrow * HD_ + ch * 32 + lhalf * 16));
      v16bf k0 = __builtin_bit_cast(
          v16bf, *(const v16us*)(kp0 + ch * 32 + lhalf * 16));
      c0 = __builtin_amdgcn_wmma_f32_16x16x32_bf16(
          false, qf, false, k0, (short)0, c0, false, false);
      v16bf k1 = __builtin_bit_cast(
          v16bf, *(const v16us*)(kp1 + ch * 32 + lhalf * 16));
      c1 = __builtin_amdgcn_wmma_f32_16x16x32_bf16(
          false, qf, false, k1, (short)0, c1, false, false);
    }

    // ---- cooperative V^T staging: Vts[d][key], 32 keys x 128 d
    // 4-key x 4-d micro-transpose per thread: 4 x global b64 in,
    // 4 x ds_store_b64 out.
    {
      const int key4 = (tid & 7) * 4;         // 0..28
      const int d4   = (tid >> 3) * 4;        // 0..124
      v4us kbuf[4];
#pragma unroll
      for (int kk = 0; kk < 4; ++kk)
        kbuf[kk] = *(const v4us*)(
            Vc + (((size_t)b * S_ + ks + key4 + kk) * NKV_ + kv) * HD_ + d4);
#pragma unroll
      for (int dd = 0; dd < 4; ++dd) {
        v4us pack = (v4us){kbuf[0][dd], kbuf[1][dd], kbuf[2][dd], kbuf[3][dd]};
        *(v4us*)(Vts + (d4 + dd) * 32 + key4) = pack;
      }
    }
    __syncthreads();

    // ---- online softmax (rows span 16-lane half groups)
    const int n = lrow;
#pragma unroll
    for (int r = 0; r < 8; ++r) {
      const int m  = r + lhalf * 8;
      const int qi = qbase + m;
      float s0 = c0[r] * scl; if (ks + n      > qi) s0 = -1e30f;
      float s1 = c1[r] * scl; if (ks + 16 + n > qi) s1 = -1e30f;
      float mx = fmaxf(s0, s1);
#pragma unroll
      for (int off = 1; off < 16; off <<= 1)
        mx = fmaxf(mx, __shfl_xor(mx, off, 16));
      const float mnew = fmaxf(mrow[r], mx);
      const float p0 = __expf(s0 - mnew);
      const float p1 = __expf(s1 - mnew);
      float ps = p0 + p1;
#pragma unroll
      for (int off = 1; off < 16; off <<= 1)
        ps += __shfl_xor(ps, off, 16);
      const float alpha = __expf(mrow[r] - mnew);
      lsum[r] = lsum[r] * alpha + ps;
      mrow[r] = mnew;
      P[m * 32 + n]      = f2bf(p0);
      P[m * 32 + 16 + n] = f2bf(p1);
#pragma unroll
      for (int dc = 0; dc < 8; ++dc) acc[dc][r] *= alpha;
    }

    // ---- PV: P (16x32) x V (32x16) for each of 8 d-chunks
    v16bf pa = __builtin_bit_cast(
        v16bf, *(const v16us*)(P + lrow * 32 + lhalf * 16));
#pragma unroll
    for (int dc = 0; dc < 8; ++dc) {
      v16bf vb = __builtin_bit_cast(
          v16bf, *(const v16us*)(Vts + (dc * 16 + lrow) * 32 + lhalf * 16));
      acc[dc] = __builtin_amdgcn_wmma_f32_16x16x32_bf16(
          false, pa, false, vb, (short)0, acc[dc], false, false);
    }
    __syncthreads();   // before next V^T staging overwrites Vts
  }

  // ---- epilogue: normalize and store O[b, s, h, d] bf16
#pragma unroll
  for (int dc = 0; dc < 8; ++dc) {
#pragma unroll
    for (int r = 0; r < 8; ++r) {
      const int m  = r + lhalf * 8;
      const int qi = qbase + m;
      const float v = acc[dc][r] / lsum[r];
      O[(((size_t)b * S_ + qi) * NH_ + h) * HD_ + dc * 16 + lrow] = f2bf(v);
    }
  }
}

// ---------------------------------------------------------------------------
// Host-side orchestration
// ---------------------------------------------------------------------------
extern "C" void kernel_launch(void* const* d_in, const int* in_sizes, int n_in,
                              void* d_out, int out_size, void* d_ws, size_t ws_size,
                              hipStream_t stream)
{
  const float* hidden = (const float*)d_in[0];
  const float* cosb   = (const float*)d_in[1];
  const float* sinb   = (const float*)d_in[2];
  const float* Wq     = (const float*)d_in[3];
  const float* Wk     = (const float*)d_in[4];
  const float* Wv     = (const float*)d_in[5];
  const float* Wo     = (const float*)d_in[6];
  const int*   idx_q  = (const int*)d_in[7];
  const int*   idx_k  = (const int*)d_in[8];
  const int*   idx_v  = (const int*)d_in[9];
  const int*   idx_o  = (const int*)d_in[10];

  const int M = B_ * S_;                  // 4096 token rows

  size_t off = 0;
  auto carve = [&](size_t bytes) {
    void* p = (char*)d_ws + off;
    off += (bytes + 255) & ~(size_t)255;
    return p;
  };
  unsigned short* Xbuf = (unsigned short*)carve((size_t)M * HID_ * 2);       // 32MB
  unsigned short* Wbuf = (unsigned short*)carve((size_t)HID_ * HID_ * 2);    // 32MB
  unsigned short* Qb   = (unsigned short*)carve((size_t)M * HID_ * 2);       // 32MB
  unsigned short* Kb   = (unsigned short*)carve((size_t)M * NKV_ * HD_ * 2); //  8MB
  unsigned short* Vb   = (unsigned short*)carve((size_t)M * NKV_ * HD_ * 2); //  8MB
  unsigned short* Ob   = (unsigned short*)carve((size_t)M * HID_ * 2);       // 32MB

  // ---- Q projection
  quant_gather<false><<<M,    256, 0, stream>>>(hidden, idx_q, Xbuf, HID_);
  quant_gather<false><<<HID_, 256, 0, stream>>>(Wq,     idx_q, Wbuf, HID_);
  gemm_bf16<false><<<dim3(M / 128, HID_ / 128), 256, 0, stream>>>(
      Xbuf, Wbuf, Qb, M, HID_, HID_);

  // ---- K projection
  quant_gather<false><<<M,          256, 0, stream>>>(hidden, idx_k, Xbuf, HID_);
  quant_gather<false><<<NKV_ * HD_, 256, 0, stream>>>(Wk,     idx_k, Wbuf, HID_);
  gemm_bf16<false><<<dim3(M / 128, (NKV_ * HD_) / 128), 256, 0, stream>>>(
      Xbuf, Wbuf, Kb, M, NKV_ * HD_, HID_);

  // ---- V projection
  quant_gather<false><<<M,          256, 0, stream>>>(hidden, idx_v, Xbuf, HID_);
  quant_gather<false><<<NKV_ * HD_, 256, 0, stream>>>(Wv,     idx_v, Wbuf, HID_);
  gemm_bf16<false><<<dim3(M / 128, (NKV_ * HD_) / 128), 256, 0, stream>>>(
      Xbuf, Wbuf, Vb, M, NKV_ * HD_, HID_);

  // ---- RoPE on Q and K
  rope_kernel<<<M, 256, 0, stream>>>(Qb, Kb, cosb, sinb);

  // ---- causal flash attention
  flash_kernel<<<dim3(S_ / 128, NH_, B_), 256, 0, stream>>>(Qb, Kb, Vb, Ob);

  // ---- output projection (fp32 result)
  quant_gather<true ><<<M,    256, 0, stream>>>(Ob, idx_o, Xbuf, HID_);
  quant_gather<false><<<HID_, 256, 0, stream>>>(Wo, idx_o, Wbuf, HID_);
  gemm_bf16<true><<<dim3(M / 128, HID_ / 128), 256, 0, stream>>>(
      Xbuf, Wbuf, (float*)d_out, M, HID_, HID_);
}
